// HetGAT_mean_76682346102830
// MI455X (gfx1250) — matrix operand order repaired and checked by reading, not compiled, not run on hardware
//
#include <hip/hip_runtime.h>
#include <hip/hip_bf16.h>
#include <math.h>

typedef _Float16 f16;
typedef _Float16 v4h  __attribute__((ext_vector_type(4)));
typedef _Float16 v8h  __attribute__((ext_vector_type(8)));
typedef _Float16 v16h __attribute__((ext_vector_type(16)));
typedef float    v4f  __attribute__((ext_vector_type(4)));
typedef float    v8f  __attribute__((ext_vector_type(8)));

// ---------------------------------------------------------------------------
// WMMA GEMM: Y[N,64] = act(X[N,K] @ W[K,64] + b)   (K = 64 or 128)
// block = 256 threads = 8 waves; each wave computes a 16x64 slab.
// X staged row-major (A frags = 2x contiguous b128), W staged column-major
// (B frags = 2x contiguous b128) -> all LDS traffic is ds_load_b128.
// ---------------------------------------------------------------------------
template<int K, int ACT>
__global__ __launch_bounds__(256)
void gemm_n64(const float* __restrict__ X, const float* __restrict__ W,
              const float* __restrict__ b, float* __restrict__ Y, int N)
{
    constexpr int XS = K + 8;    // halves; (K+8)*2 bytes is a multiple of 16
    constexpr int WS = K + 8;    // column pitch for transposed W
    __shared__ f16 sX[128 * XS];
    __shared__ f16 sW[64 * WS];

    const int tid  = threadIdx.x;
    const int row0 = blockIdx.x * 128;

    // stage X tile (128 rows x K), fp32 -> f16, vectorized 4-wide
    constexpr int KV = K / 4;
    for (int idx = tid; idx < 128 * KV; idx += 256) {
        int r = idx / KV, c4 = idx - r * KV;
        int gr = row0 + r;
        v4f v = {};
        if (gr < N) v = *(const v4f*)(X + (size_t)gr * K + c4 * 4);
        v4h h;
        h[0] = (f16)v[0]; h[1] = (f16)v[1]; h[2] = (f16)v[2]; h[3] = (f16)v[3];
        *(v4h*)(&sX[r * XS + c4 * 4]) = h;
    }
    // stage W (K x 64) transposed into sW[n * WS + k], fp32 -> f16
    for (int idx = tid; idx < K * 16; idx += 256) {
        int k = idx >> 4, n4 = (idx & 15) * 4;
        v4f v = *(const v4f*)(W + (size_t)k * 64 + n4);
        sW[(n4 + 0) * WS + k] = (f16)v[0];
        sW[(n4 + 1) * WS + k] = (f16)v[1];
        sW[(n4 + 2) * WS + k] = (f16)v[2];
        sW[(n4 + 3) * WS + k] = (f16)v[3];
    }
    __syncthreads();

    const int wave = tid >> 5;
    const int lane = tid & 31;
    const int m    = lane & 15;   // row (A) or col (B/D) within 16
    const int hh   = lane >> 4;   // lane half
    const int wrow = wave * 16;

    v8f acc[4] = {};

    for (int k0 = 0; k0 < K; k0 += 32) {
        // A fragment 16x32 f16: two contiguous 16B runs per lane
        const f16* pA = &sX[(wrow + m) * XS + k0];
        v8h alo = *(const v8h*)(pA + hh * 8);
        v8h ahi = *(const v8h*)(pA + 16 + hh * 8);
        v16h a = __builtin_shufflevector(alo, ahi,
                    0, 1, 2, 3, 4, 5, 6, 7, 8, 9, 10, 11, 12, 13, 14, 15);
        #pragma unroll
        for (int t = 0; t < 4; ++t) {
            const int n0 = t * 16;
            // B fragment 32x16 f16: two contiguous 16B runs per lane
            const f16* pB = &sW[(n0 + m) * WS + k0];
            v8h blo = *(const v8h*)(pB + hh * 8);
            v8h bhi = *(const v8h*)(pB + 16 + hh * 8);
            v16h bb = __builtin_shufflevector(blo, bhi,
                    0, 1, 2, 3, 4, 5, 6, 7, 8, 9, 10, 11, 12, 13, 14, 15);
            acc[t] = __builtin_amdgcn_wmma_f32_16x16x32_f16(
                false, a, false, bb, (short)0, acc[t], false, false);
        }
    }

    // store D: VGPR r holds row (r + hh*8), col = lane&15 within tile
    #pragma unroll
    for (int t = 0; t < 4; ++t) {
        const int n = t * 16 + m;
        const float bias = b[n];
        #pragma unroll
        for (int r = 0; r < 8; ++r) {
            int gr = row0 + wrow + r + hh * 8;
            if (gr < N) {
                float v = acc[t][r] + bias;
                if (ACT == 1) v = v > 0.0f ? v : 0.0f;
                Y[(size_t)gr * 64 + n] = v;
            }
        }
    }
}

// ---------------------------------------------------------------------------
// Per-node attention scalars: wave per node, lanes cover 64 channels.
// out_x1 = x.a1 ; out_h1 = x.a2 ; out_w2 = exp(lrelu(x.a1 + x.a2))
// ---------------------------------------------------------------------------
__global__ __launch_bounds__(256)
void node_dots(const float* __restrict__ x, const float* __restrict__ a1,
               const float* __restrict__ a2, float* __restrict__ out_x1,
               float* __restrict__ out_w2, float* __restrict__ out_h1, int N)
{
    int wave = (blockIdx.x * blockDim.x + threadIdx.x) >> 5;
    int lane = threadIdx.x & 31;
    if (wave >= N) return;
    float v0 = x[(size_t)wave * 64 + lane];
    float v1 = x[(size_t)wave * 64 + lane + 32];
    float d1 = v0 * a1[lane] + v1 * a1[lane + 32];
    float d2 = v0 * a2[lane] + v1 * a2[lane + 32];
    #pragma unroll
    for (int off = 16; off > 0; off >>= 1) {
        d1 += __shfl_xor(d1, off, 32);
        d2 += __shfl_xor(d2, off, 32);
    }
    if (lane == 0) {
        if (out_x1) out_x1[wave] = d1;
        if (out_h1) out_h1[wave] = d2;
        if (out_w2) {
            float s = d1 + d2;
            float l = s >= 0.0f ? s : 0.2f * s;
            out_w2[wave] = expf(l);
        }
    }
}

// ---------------------------------------------------------------------------
// Edge scatter: wave per edge. w = exp(lrelu(x1[s]+h1[t]));
// num[s,:] += w*h[t,:]; div[s] += w.  Coalesced 64-ch gather, f32 atomics.
// ---------------------------------------------------------------------------
__global__ __launch_bounds__(256)
void edge_scatter(const int* __restrict__ src, const int* __restrict__ tgt,
                  const float* __restrict__ x1s, const float* __restrict__ h1t,
                  const float* __restrict__ h, float* __restrict__ num,
                  float* __restrict__ divb, int E)
{
    int wave = (blockIdx.x * blockDim.x + threadIdx.x) >> 5;
    int lane = threadIdx.x & 31;
    if (wave >= E) return;
    int s = src[wave], t = tgt[wave];
    float z = x1s[s] + h1t[t];
    float l = z >= 0.0f ? z : 0.2f * z;
    float w = expf(l);
    atomicAdd(&num[(size_t)s * 64 + lane],      w * h[(size_t)t * 64 + lane]);
    atomicAdd(&num[(size_t)s * 64 + lane + 32], w * h[(size_t)t * 64 + lane + 32]);
    if (lane == 0) atomicAdd(&divb[s], w);
}

// out[i] (=|+=) (num[i] + w2[n]*x[i]) / (div[n] + w2[n])
template<int ADD>
__global__ __launch_bounds__(256)
void finalize_k(const float* __restrict__ num, const float* __restrict__ divb,
                const float* __restrict__ w2, const float* __restrict__ x,
                float* __restrict__ out, int NE)
{
    int i = blockIdx.x * blockDim.x + threadIdx.x;
    if (i >= NE) return;
    int n = i >> 6;
    float wv = w2[n];
    float val = (num[i] + wv * x[i]) / (divb[n] + wv);
    if (ADD) out[i] += val; else out[i] = val;
}

__global__ __launch_bounds__(256)
void elu_scale_k(float* __restrict__ buf, float scale, int NE)
{
    int i = blockIdx.x * blockDim.x + threadIdx.x;
    if (i >= NE) return;
    float v = buf[i] * scale;
    buf[i] = v > 0.0f ? v : (expf(v) - 1.0f);
}

__global__ __launch_bounds__(256)
void fill0_k(float* __restrict__ p, int n)
{
    int i = blockIdx.x * blockDim.x + threadIdx.x;
    if (i < n) p[i] = 0.0f;
}

// ---------------------------------------------------------------------------
extern "C" void kernel_launch(void* const* d_in, const int* in_sizes, int n_in,
                              void* d_out, int out_size, void* d_ws, size_t ws_size,
                              hipStream_t stream)
{
    const float* x_paper   = (const float*)d_in[0];
    const float* x_author  = (const float*)d_in[1];
    const float* W1_paper  = (const float*)d_in[2];
    const float* b1_paper  = (const float*)d_in[3];
    const float* W1_author = (const float*)d_in[4];
    const float* b1_author = (const float*)d_in[5];
    const float* Wfcs      = (const float*)d_in[6];
    const float* bfcs      = (const float*)d_in[7];
    const float* attn1     = (const float*)d_in[8];
    const float* attn2     = (const float*)d_in[9];
    const float* W2        = (const float*)d_in[10];
    const float* b2        = (const float*)d_in[11];
    const int*   edge_pa   = (const int*)d_in[12];
    const int*   edge_ap   = (const int*)d_in[13];
    const int*   edge_pp   = (const int*)d_in[14];

    const int NP = in_sizes[0] / 128;
    const int NA = in_sizes[1] / 128;
    const int E  = in_sizes[12] / 2;
    const int NMAX = NP > NA ? NP : NA;

    // workspace layout (floats)
    float* ws  = (float*)d_ws;
    float* xp  = ws;               ws += (size_t)NP * 64;
    float* xa  = ws;               ws += (size_t)NA * 64;
    float* yp  = ws;               ws += (size_t)NP * 64;
    float* ya  = ws;               ws += (size_t)NA * 64;
    float* num = ws;               ws += (size_t)NMAX * 64;
    float* divb= ws;               ws += (size_t)NMAX;
    float* x1s = ws;               ws += (size_t)NMAX;
    float* h1t = ws;               ws += (size_t)NMAX;
    float* w2  = ws;               ws += (size_t)NMAX;

    const dim3 blk(256);
    auto gemmGrid = [](int n) { return dim3((n + 127) / 128); };
    auto waveGrid = [](int n) { return dim3((n + 7) / 8); };
    auto elemGrid = [](int n) { return dim3((n + 255) / 256); };

    // input projection with relu (K=128)
    gemm_n64<128, 1><<<gemmGrid(NP), blk, 0, stream>>>(x_paper,  W1_paper,  b1_paper,  xp, NP);
    gemm_n64<128, 1><<<gemmGrid(NA), blk, 0, stream>>>(x_author, W1_author, b1_author, xa, NA);

    for (int i = 0; i < 2; ++i) {
        const float* Wfc = Wfcs + (size_t)i * 64 * 64;
        const float* bfc = bfcs + (size_t)i * 64;
        gemm_n64<64, 0><<<gemmGrid(NP), blk, 0, stream>>>(xp, Wfc, bfc, yp, NP);
        gemm_n64<64, 0><<<gemmGrid(NA), blk, 0, stream>>>(xa, Wfc, bfc, ya, NA);

        // --- edge type 0: paper->author (aggregate onto paper; x=yp, h=ya) ---
        {
            const float* a1 = attn1 + (size_t)(i * 3 + 0) * 64;
            const float* a2 = attn2 + (size_t)(i * 3 + 0) * 64;
            fill0_k<<<elemGrid(NP * 64), blk, 0, stream>>>(num, NP * 64);
            fill0_k<<<elemGrid(NP), blk, 0, stream>>>(divb, NP);
            node_dots<<<waveGrid(NP), blk, 0, stream>>>(yp, a1, a2, x1s, w2, nullptr, NP);
            node_dots<<<waveGrid(NA), blk, 0, stream>>>(ya, a1, a2, nullptr, nullptr, h1t, NA);
            edge_scatter<<<waveGrid(E), blk, 0, stream>>>(edge_pa, edge_pa + E, x1s, h1t, ya, num, divb, E);
            finalize_k<0><<<elemGrid(NP * 64), blk, 0, stream>>>(num, divb, w2, yp, xp, NP * 64);
        }
        // --- edge type 2: paper->paper (x = h = yp) ---
        {
            const float* a1 = attn1 + (size_t)(i * 3 + 2) * 64;
            const float* a2 = attn2 + (size_t)(i * 3 + 2) * 64;
            fill0_k<<<elemGrid(NP * 64), blk, 0, stream>>>(num, NP * 64);
            fill0_k<<<elemGrid(NP), blk, 0, stream>>>(divb, NP);
            node_dots<<<waveGrid(NP), blk, 0, stream>>>(yp, a1, a2, x1s, w2, h1t, NP);
            edge_scatter<<<waveGrid(E), blk, 0, stream>>>(edge_pp, edge_pp + E, x1s, h1t, yp, num, divb, E);
            finalize_k<1><<<elemGrid(NP * 64), blk, 0, stream>>>(num, divb, w2, yp, xp, NP * 64);
        }
        // paper: mean of 2 edge types + ELU
        elu_scale_k<<<elemGrid(NP * 64), blk, 0, stream>>>(xp, 0.5f, NP * 64);

        // --- edge type 1: author->paper (aggregate onto author; x=ya, h=yp) ---
        {
            const float* a1 = attn1 + (size_t)(i * 3 + 1) * 64;
            const float* a2 = attn2 + (size_t)(i * 3 + 1) * 64;
            fill0_k<<<elemGrid(NA * 64), blk, 0, stream>>>(num, NA * 64);
            fill0_k<<<elemGrid(NA), blk, 0, stream>>>(divb, NA);
            node_dots<<<waveGrid(NA), blk, 0, stream>>>(ya, a1, a2, x1s, w2, nullptr, NA);
            node_dots<<<waveGrid(NP), blk, 0, stream>>>(yp, a1, a2, nullptr, nullptr, h1t, NP);
            edge_scatter<<<waveGrid(E), blk, 0, stream>>>(edge_ap, edge_ap + E, x1s, h1t, yp, num, divb, E);
            finalize_k<0><<<elemGrid(NA * 64), blk, 0, stream>>>(num, divb, w2, ya, xa, NA * 64);
        }
        elu_scale_k<<<elemGrid(NA * 64), blk, 0, stream>>>(xa, 1.0f, NA * 64);
    }

    // final projection: out = xp @ W2 + b2
    gemm_n64<64, 0><<<gemmGrid(NP), blk, 0, stream>>>(xp, W2, b2, (float*)d_out, NP);
}
